// LoRALayer_53618371723762
// MI455X (gfx1250) — compile-verified, matrix operand order checked by reading, not run on hardware
//
#include <hip/hip_runtime.h>

// ---------------------------------------------------------------------------
// LoRA layer for MI455X (gfx1250):  out = x @ (W + alpha*B@A)^T + bias
// M=16384 (B*S), N=4096 (D_out), K=4096 (D_in), R=16, fp32 in/out.
//   Kernel 1: fold rank-16 update into bf16 effective weight  (ws, 32 MB)
//   Kernel 2: convert x fp32 -> bf16 once                      (ws, 128 MB)
//   Kernel 3: bf16 WMMA GEMM, tiles staged by the Tensor Data Mover with
//             double-buffered LDS + s_wait_tensorcnt pipelining.
// ---------------------------------------------------------------------------

typedef __attribute__((ext_vector_type(16))) __bf16        v16bf;
typedef __attribute__((ext_vector_type(8)))  float         v8f;
typedef __attribute__((ext_vector_type(4)))  unsigned int  v4u;
typedef __attribute__((ext_vector_type(8)))  int           v8i_t;
typedef __attribute__((ext_vector_type(4)))  int           v4i_t;

#define D_IN   4096
#define D_OUT  4096
#define RANK   16
#define M_TOT  16384
#define ALPHA  16.0f

#define BM 128
#define BN 128
#define BK 32
#define LDT 40   // padded LDS row in ushorts: 64B data + 16B pad = 80B (16B aligned)

#if __has_builtin(__builtin_amdgcn_tensor_load_to_lds) && __has_builtin(__builtin_amdgcn_s_wait_tensorcnt)
#define USE_TDM 1
#else
#define USE_TDM 0
#endif

// round-to-nearest-even fp32 -> bf16 bits
static __device__ __forceinline__ unsigned f2bf(float f) {
    union { float f; unsigned u; } c; c.f = f;
    unsigned r = c.u + 0x7FFFu + ((c.u >> 16) & 1u);
    return r >> 16;
}

// byte offset of a __shared__ object inside LDS (addrspace(3) ptrtoint)
static __device__ __forceinline__ unsigned lds_off(const void* p) {
    return (unsigned)(unsigned long long)(__attribute__((address_space(3))) const char*)p;
}

// ---------------------------------------------------------------------------
// Kernel 1: weff[o][i] = bf16( W[o][i] + ALPHA * sum_r B[o][r]*A[r][i] )
// ---------------------------------------------------------------------------
__global__ void lora_prep_weff(const float* __restrict__ W,
                               const float* __restrict__ A,
                               const float* __restrict__ Bm,
                               unsigned short* __restrict__ weff) {
    const int o  = blockIdx.y;
    const int i0 = (blockIdx.x * 256 + threadIdx.x) * 8;

    float br[RANK];
#pragma unroll
    for (int r = 0; r < RANK; ++r) br[r] = Bm[o * RANK + r];   // uniform -> s_load

    float acc[8];
#pragma unroll
    for (int j = 0; j < 8; ++j) acc[j] = 0.0f;

#pragma unroll
    for (int r = 0; r < RANK; ++r) {
        const float4* ap = (const float4*)&A[r * D_IN + i0];
        float4 a0 = ap[0], a1 = ap[1];
        acc[0] += br[r] * a0.x; acc[1] += br[r] * a0.y;
        acc[2] += br[r] * a0.z; acc[3] += br[r] * a0.w;
        acc[4] += br[r] * a1.x; acc[5] += br[r] * a1.y;
        acc[6] += br[r] * a1.z; acc[7] += br[r] * a1.w;
    }

    const float4* wp = (const float4*)&W[o * D_IN + i0];
    float4 w0 = wp[0], w1 = wp[1];

    uint4 pk;
    pk.x = f2bf(w0.x + ALPHA * acc[0]) | (f2bf(w0.y + ALPHA * acc[1]) << 16);
    pk.y = f2bf(w0.z + ALPHA * acc[2]) | (f2bf(w0.w + ALPHA * acc[3]) << 16);
    pk.z = f2bf(w1.x + ALPHA * acc[4]) | (f2bf(w1.y + ALPHA * acc[5]) << 16);
    pk.w = f2bf(w1.z + ALPHA * acc[6]) | (f2bf(w1.w + ALPHA * acc[7]) << 16);
    *(uint4*)&weff[o * D_IN + i0] = pk;
}

// ---------------------------------------------------------------------------
// Kernel 2: xbf = bf16(x), one pass (8 elements / thread)
// ---------------------------------------------------------------------------
__global__ void lora_prep_xbf(const float* __restrict__ x,
                              unsigned short* __restrict__ xbf) {
    const size_t i0 = ((size_t)blockIdx.x * 256 + threadIdx.x) * 8;
    const float4* xp = (const float4*)&x[i0];
    float4 a = xp[0], b = xp[1];
    uint4 pk;
    pk.x = f2bf(a.x) | (f2bf(a.y) << 16);
    pk.y = f2bf(a.z) | (f2bf(a.w) << 16);
    pk.z = f2bf(b.x) | (f2bf(b.y) << 16);
    pk.w = f2bf(b.z) | (f2bf(b.w) << 16);
    *(uint4*)&xbf[i0] = pk;
}

// ---------------------------------------------------------------------------
// TDM: issue a 2D tile load (rows x 32 bf16) into LDS with 16B row padding.
// D# layout per CDNA5 ISA ch.8 (async_tensor): group0 = {flags, lds, gaddr},
// group1 = {dsize/pad, tensor dims, tile dims, stride}. Groups 2/3 unused
// for a 2D tile -> zero-filled.
// ---------------------------------------------------------------------------
#if USE_TDM
static __device__ __forceinline__ void tdm_load_tile(const unsigned short* gptr,
                                                     unsigned lds_byte,
                                                     int rows, int tensor_rows) {
    unsigned long long ga = (unsigned long long)gptr;
    const unsigned ga_lo = __builtin_amdgcn_readfirstlane((unsigned)ga);
    const unsigned ga_hi = __builtin_amdgcn_readfirstlane((unsigned)(ga >> 32));
    const unsigned lds   = __builtin_amdgcn_readfirstlane(lds_byte);

    v4u g0;
    g0[0] = 1u;                                        // count=1, user mode
    g0[1] = lds;                                       // lds_addr (bytes)
    g0[2] = ga_lo;                                     // global_addr[31:0]
    g0[3] = (ga_hi & 0x01FFFFFFu) | (2u << 30);        // addr[56:32] | type=2

    v8i_t g1;
    g1[0] = (int)((1u << 16)      // data_size = 2 bytes
                | (1u << 20)      // pad_enable
                | (3u << 22)      // pad_interval: every 16 DWORDs (64B row)
                | (3u << 25));    // pad_amount: 4 DWORDs (16B)
    g1[1] = (int)(((unsigned)D_IN & 0xFFFFu) << 16);                       // tensor_dim0 lo
    g1[2] = (int)((((unsigned)D_IN >> 16) & 0xFFFFu)
                | (((unsigned)tensor_rows & 0xFFFFu) << 16));              // dim0 hi | dim1 lo
    g1[3] = (int)((((unsigned)tensor_rows >> 16) & 0xFFFFu)
                | ((unsigned)BK << 16));                                   // dim1 hi | tile_dim0
    g1[4] = rows;                                                          // tile_dim1 (tile_dim2=0)
    g1[5] = D_IN;                                                          // tensor_dim0_stride lo
    g1[6] = 0;
    g1[7] = 0;

    const v4i_t z4 = (v4i_t){0, 0, 0, 0};
    const v8i_t z8 = (v8i_t){0, 0, 0, 0, 0, 0, 0, 0};
    // clang-23 / therock-10.0 arity: (g0, g1, g2, g3, g_extra, cpol)
    __builtin_amdgcn_tensor_load_to_lds(g0, g1, z4, z4, z8, 0);
}
#endif

// ---------------------------------------------------------------------------
// Kernel 3: out[m][n] = sum_k xbf[m][k]*weff[n][k] + bias[n]
// 128x128 tile / block, BK=32, 256 threads = 8 waves (2M x 4N).
// Each wave: 4x2 WMMA tiles -> 8x v_wmma_f32_16x16x32_bf16 / K-step.
// ---------------------------------------------------------------------------
__global__ void __launch_bounds__(256)
lora_gemm_bf16(const unsigned short* __restrict__ xbf,
               const unsigned short* __restrict__ weff,
               const float* __restrict__ bias,
               float* __restrict__ out) {
    __shared__ __attribute__((aligned(16))) unsigned short As[2][BM][LDT];
    __shared__ __attribute__((aligned(16))) unsigned short Bs[2][BN][LDT];

    const int tid   = threadIdx.x;
    const int lane  = tid & 31;
    const int wave  = tid >> 5;          // 0..7
    const int waveM = wave & 1;          // 2 waves along M (64 rows each)
    const int waveN = wave >> 1;         // 4 waves along N (32 cols each)
    const int half  = lane >> 4;         // K-half selector
    const int lr    = lane & 15;         // row (A) / col (B) within 16-tile

    const int m0 = blockIdx.y * BM;
    const int n0 = blockIdx.x * BN;

    v8f acc[4][2];
#pragma unroll
    for (int mt = 0; mt < 4; ++mt)
#pragma unroll
        for (int nt = 0; nt < 2; ++nt)
            acc[mt][nt] = (v8f){0.f, 0.f, 0.f, 0.f, 0.f, 0.f, 0.f, 0.f};

#if USE_TDM
    // ---- prologue: DMA tile 0 into buffer 0 ----
    if (wave == 0) {
        tdm_load_tile(&xbf[(size_t)m0 * D_IN], lds_off(&As[0][0][0]), BM, M_TOT);
        tdm_load_tile(&weff[(size_t)n0 * D_IN], lds_off(&Bs[0][0][0]), BN, D_OUT);
    }
#endif

    int cur = 0;
    for (int k0 = 0; k0 < D_IN; k0 += BK) {
        const bool has_next = (k0 + BK < D_IN);

#if USE_TDM
        if (wave == 0) {
            if (has_next) {   // DMA next tile into the other buffer
                tdm_load_tile(&xbf[(size_t)m0 * D_IN + k0 + BK],
                              lds_off(&As[cur ^ 1][0][0]), BM, M_TOT);
                tdm_load_tile(&weff[(size_t)n0 * D_IN + k0 + BK],
                              lds_off(&Bs[cur ^ 1][0][0]), BN, D_OUT);
                __builtin_amdgcn_s_wait_tensorcnt(2);   // current tile complete
            } else {
                __builtin_amdgcn_s_wait_tensorcnt(0);
            }
        }
        __syncthreads();                                 // tile `cur` visible to all
#else
        // fallback: cooperative staging (both operands already bf16)
        {
            const int row = tid >> 2, cq = (tid & 3) * 8;   // 4 uint4 per 32-elem row
#pragma unroll
            for (int p = 0; p < 2; ++p) {
                const int r = p * 64 + row;
                *(uint4*)&As[0][r][cq] = *(const uint4*)&xbf[(size_t)(m0 + r) * D_IN + k0 + cq];
                *(uint4*)&Bs[0][r][cq] = *(const uint4*)&weff[(size_t)(n0 + r) * D_IN + k0 + cq];
            }
        }
        __syncthreads();
        cur = 0;
#endif

        // ---- fragments: 16 contiguous bf16 per lane (2x ds_load_b128) ----
        union Frag { uint4 q[2]; v16bf v; };
        Frag a[4], b[2];
#pragma unroll
        for (int mt = 0; mt < 4; ++mt) {
            const unsigned short* p = &As[cur][waveM * 64 + mt * 16 + lr][half * 16];
            a[mt].q[0] = *(const uint4*)p;
            a[mt].q[1] = *(const uint4*)(p + 8);
        }
#pragma unroll
        for (int nt = 0; nt < 2; ++nt) {
            const unsigned short* p = &Bs[cur][waveN * 32 + nt * 16 + lr][half * 16];
            b[nt].q[0] = *(const uint4*)p;
            b[nt].q[1] = *(const uint4*)(p + 8);
        }

#pragma unroll
        for (int mt = 0; mt < 4; ++mt)
#pragma unroll
            for (int nt = 0; nt < 2; ++nt)
                acc[mt][nt] = __builtin_amdgcn_wmma_f32_16x16x32_bf16(
                    false, a[mt].v, false, b[nt].v,
                    (short)0, acc[mt][nt], false, false);

        __syncthreads();   // all reads of `cur` done before TDM overwrites it
        cur ^= 1;
    }

    // ---- epilogue: C/D layout is VGPR i, lane-half -> row, lane%16 -> col ----
#pragma unroll
    for (int nt = 0; nt < 2; ++nt) {
        const int gn = n0 + waveN * 32 + nt * 16 + lr;
        const float bv = bias[gn];
#pragma unroll
        for (int mt = 0; mt < 4; ++mt) {
            const int gmb = m0 + waveM * 64 + mt * 16 + half * 8;
#pragma unroll
            for (int i = 0; i < 8; ++i)
                out[(size_t)(gmb + i) * D_OUT + gn] = acc[mt][nt][i] + bv;
        }
    }
}

// ---------------------------------------------------------------------------
extern "C" void kernel_launch(void* const* d_in, const int* in_sizes, int n_in,
                              void* d_out, int out_size, void* d_ws, size_t ws_size,
                              hipStream_t stream) {
    const float* x    = (const float*)d_in[0];
    const float* W    = (const float*)d_in[1];
    const float* A    = (const float*)d_in[2];
    const float* Bm   = (const float*)d_in[3];
    const float* bias = (const float*)d_in[4];
    float* out = (float*)d_out;

    unsigned short* weff = (unsigned short*)d_ws;                       // 32 MB
    unsigned short* xbf  = weff + (size_t)D_OUT * D_IN;                 // 128 MB

    dim3 gPrep(D_IN / (256 * 8), D_OUT);                                // (2, 4096)
    lora_prep_weff<<<gPrep, 256, 0, stream>>>(W, A, Bm, weff);

    const size_t xElems = (size_t)M_TOT * D_IN;
    lora_prep_xbf<<<dim3((unsigned)(xElems / (256 * 8))), 256, 0, stream>>>(x, xbf);

    dim3 gGemm(D_OUT / BN, M_TOT / BM);                                 // (32, 128)
    lora_gemm_bf16<<<gGemm, 256, 0, stream>>>(xbf, weff, bias, out);
}
